// SparseAttention_26113401160135
// MI455X (gfx1250) — compile-verified, matrix operand order checked by reading, not run on hardware
//
#include <hip/hip_runtime.h>
#include <hip/hip_bf16.h>
#include <cstdint>

// ---------------------------------------------------------------------------
// CDNA5 (gfx1250) sparse attention.
//  B=4, N=4096, d_model=256, d_k=d_v=64, top-k = 2048 (SPARSITY 0.5).
//
//  K1 qkv_proj : x[16384,256] @ Wqkv[256,192] + b -> q,k (bf16 row-major),
//                V stored TRANSPOSED as Vt[b][64][4096] bf16 (so PV B-frags
//                are contiguous b128 loads).
//  K2 attn     : per 16-query block: S=QK^T (bf16 WMMA) into 256KB LDS,
//                exact 2048-th-largest threshold via 4-pass radix select,
//                masked softmax (mask zeros participate, per reference),
//                O = P@V (bf16 WMMA) -> attn_out bf16 (ws).
//  K3 out_proj : attn_out[16384,64] @ Wout[64,256] + b -> d_out f32.
// ---------------------------------------------------------------------------

typedef __attribute__((ext_vector_type(16))) __bf16 v16bf;
typedef __attribute__((ext_vector_type(8)))  __bf16 v8bf;
typedef __attribute__((ext_vector_type(8)))  float  v8f;
typedef __attribute__((ext_vector_type(4)))  float  v4f;

#define DEV __device__ __forceinline__

DEV __bf16 tobf(float f) { return (__bf16)f; }

DEV v16bf cat8(v8bf a, v8bf b) {
  return __builtin_shufflevector(a, b, 0,1,2,3,4,5,6,7,8,9,10,11,12,13,14,15);
}

DEV v8f wmma_bf16(v16bf a, v16bf b, v8f c) {
  // D = A(16x32 bf16) * B(32x16 bf16) + C(16x16 f32)
  return __builtin_amdgcn_wmma_f32_16x16x32_bf16(
      /*neg_a=*/false, a, /*neg_b=*/false, b,
      /*c_mod=*/(short)0, c, /*reuse_a=*/false, /*reuse_b=*/false);
}

// Monotonic float <-> orderable uint mapping (ascending).
DEV uint32_t f2ord(float f) {
  uint32_t u = __float_as_uint(f);
  return (u & 0x80000000u) ? ~u : (u | 0x80000000u);
}
DEV float ord2f(uint32_t k) {
  uint32_t u = (k & 0x80000000u) ? (k & 0x7FFFFFFFu) : ~k;
  return __uint_as_float(u);
}

static constexpr int ROWS   = 4 * 4096;   // B*N
static constexpr int DM     = 256;        // d_model
static constexpr int DQKV   = 192;        // 2*d_k + d_v
static constexpr int DK     = 64;
static constexpr int NKEY   = 4096;
static constexpr int TOPK   = 2048;

// LDS layout for attention kernel (bytes)
static constexpr int LDS_S     = 16 * NKEY * 4;     // 262144: score stripe
static constexpr int LDS_HIST  = 8 * 256 * 4;       //   8192: per-wave histograms
static constexpr int LDS_THR   = 16 * 4;
static constexpr int LDS_INV   = 16 * 4;
static constexpr int LDS_PART  = 8 * 16 * 64 * 4;   //  32768: PV partials
static constexpr int LDS_TOTAL = LDS_S + LDS_HIST + LDS_THR + LDS_INV + LDS_PART;

// ---------------------------------------------------------------------------
// K1: qkv projection. One wave per 16x16 output tile; K=256 in 8 WMMA steps.
// Tile-uniform output routing: col tiles 0-3 -> q, 4-7 -> k, 8-11 -> Vt.
// ---------------------------------------------------------------------------
__global__ __launch_bounds__(32)
void qkv_proj_kernel(const float* __restrict__ x, const float* __restrict__ Wqkv,
                     const float* __restrict__ bqkv,
                     __bf16* __restrict__ gq, __bf16* __restrict__ gk,
                     __bf16* __restrict__ gvt)
{
  const int lane = threadIdx.x;
  const int row0 = blockIdx.x * 16;           // over 16384
  const int col0 = blockIdx.y * 16;           // over 192
  const int ln   = lane & 15;
  const int half = lane >> 4;

  v8f acc = {};
  const float* xr = x + (size_t)(row0 + ln) * DM;
  for (int k0 = 0; k0 < DM; k0 += 32) {
    const int ka = k0 + half * 8;
    const int kc = k0 + 16 + half * 8;
    v16bf A, Bv;
#pragma unroll
    for (int j = 0; j < 8; ++j) {
      A[j]     = tobf(xr[ka + j]);
      A[8 + j] = tobf(xr[kc + j]);
      Bv[j]     = tobf(Wqkv[(size_t)(ka + j) * DQKV + col0 + ln]);
      Bv[8 + j] = tobf(Wqkv[(size_t)(kc + j) * DQKV + col0 + ln]);
    }
    acc = wmma_bf16(A, Bv, acc);
  }

  const float bias = bqkv[col0 + ln];         // per-lane column bias
  if (col0 < 128) {
    // q or k, row-major [16384][64]; block-uniform select.
    __bf16* dst = (col0 < DK) ? gq : gk;
    const int c = (col0 < DK) ? col0 : (col0 - DK);
#pragma unroll
    for (int r = 0; r < 8; ++r) {
      const int mm = r + 8 * half;
      dst[(size_t)(row0 + mm) * DK + c + ln] = tobf(acc[r] + bias);
    }
  } else {
    // V transposed: Vt[bat][64][4096]. Lane's 8 rows are 8 consecutive keys
    // at fixed column -> one 16-byte store.
    const int bat = row0 >> 12;
    const int keybase = (row0 & (NKEY - 1)) + 8 * half;
    const int c = col0 - 128;
    v8bf pk;
#pragma unroll
    for (int r = 0; r < 8; ++r) pk[r] = tobf(acc[r] + bias);
    *(v8bf*)(gvt + ((size_t)bat * DK + c + ln) * NKEY + keybase) = pk;
  }
}

// ---------------------------------------------------------------------------
// K2: attention. One workgroup (8 wave32) per (batch, 16-query block).
// ---------------------------------------------------------------------------
__global__ __launch_bounds__(256)
void attention_kernel(const __bf16* __restrict__ gq, const __bf16* __restrict__ gk,
                      const __bf16* __restrict__ gvt, __bf16* __restrict__ gao)
{
  extern __shared__ char smem[];
  float*    S       = (float*)smem;                           // [16][4096]
  unsigned* hist    = (unsigned*)(smem + LDS_S);              // [8][256]
  float*    thrv    = (float*)(smem + LDS_S + LDS_HIST);      // [16]
  float*    invs    = thrv + 16;                              // [16]
  float*    partial = (float*)(smem + LDS_S + LDS_HIST + LDS_THR + LDS_INV);

  const int tid  = threadIdx.x;
  const int lane = tid & 31;
  const int wave = tid >> 5;
  const int bat  = blockIdx.x >> 8;          // 256 query blocks per batch
  const int qblk = blockIdx.x & 255;
  const size_t qrow0 = (size_t)bat * NKEY + (size_t)qblk * 16;
  const size_t kbase = (size_t)bat * NKEY;
  const int ln   = lane & 15;
  const int half = lane >> 4;

  // ---- Phase 1: S = (Q K^T) * rsqrt(d_k) into LDS -------------------------
  v16bf aq[2];
  {
    const __bf16* qr = gq + (qrow0 + ln) * DK;
#pragma unroll
    for (int s = 0; s < 2; ++s) {
      const int ka = s * 32 + half * 8;
      const int kc = s * 32 + 16 + half * 8;
      aq[s] = cat8(*(const v8bf*)(qr + ka), *(const v8bf*)(qr + kc));
    }
  }
  for (int t = wave; t < NKEY / 16; t += 8) {
    const int key0 = t * 16;
    if (t + 8 < NKEY / 16)
      __builtin_prefetch(gk + (kbase + (size_t)(key0 + 128) + ln) * DK, 0, 1);
    const __bf16* kr = gk + (kbase + key0 + ln) * DK;
    v8f acc = {};
#pragma unroll
    for (int s = 0; s < 2; ++s) {
      const int ka = s * 32 + half * 8;
      const int kc = s * 32 + 16 + half * 8;
      const v16bf Bv = cat8(*(const v8bf*)(kr + ka), *(const v8bf*)(kr + kc));
      acc = wmma_bf16(aq[s], Bv, acc);
    }
#pragma unroll
    for (int r = 0; r < 8; ++r) {
      const int mm = r + 8 * half;
      S[mm * NKEY + key0 + ln] = acc[r] * 0.125f;  // 1/sqrt(64)
    }
  }
  __syncthreads();

  // ---- Phase 2: exact 2048-th-largest per row (radix select, 4x8 bits) ----
  unsigned* myh = hist + wave * 256;
#pragma unroll 1
  for (int rr = 0; rr < 2; ++rr) {
    const int row = wave * 2 + rr;
    const float* Sr = S + row * NKEY;
    unsigned prefix = 0;
    unsigned kt = TOPK;
#pragma unroll 1
    for (int pass = 0; pass < 4; ++pass) {
      const int shift = 24 - 8 * pass;
      for (int i = lane; i < 256; i += 32) myh[i] = 0;
      __syncthreads();
      const unsigned hm = (pass == 0) ? 0u : (0xFFFFFFFFu << (shift + 8));
      for (int i = lane * 4; i < NKEY; i += 128) {
        const v4f v = *(const v4f*)(Sr + i);
#pragma unroll
        for (int j = 0; j < 4; ++j) {
          const unsigned key = f2ord(v[j]);
          if (((key ^ prefix) & hm) == 0u)
            atomicAdd(&myh[(key >> shift) & 0xFFu], 1u);
        }
      }
      __syncthreads();
      unsigned cum = 0; int sel = 0;              // identical walk in all lanes
      for (int bn = 255; bn >= 0; --bn) {
        const unsigned c = myh[bn];
        if (cum + c >= kt) { sel = bn; break; }
        cum += c;
      }
      kt -= cum;
      prefix |= (unsigned)sel << shift;
      __syncthreads();
    }
    if (lane == 0) thrv[row] = ord2f(prefix);
  }
  __syncthreads();

  // ---- Phase 3: softmax of (S * mask); masked entries contribute exp(0) ---
#pragma unroll 1
  for (int rr = 0; rr < 2; ++rr) {
    const int row = wave * 2 + rr;
    float* Sr = S + row * NKEY;
    const float thr = thrv[row];
    float mx = 0.0f;                              // zeros from mask are present
    for (int i = lane * 4; i < NKEY; i += 128) {
      const v4f v = *(const v4f*)(Sr + i);
#pragma unroll
      for (int j = 0; j < 4; ++j) mx = fmaxf(mx, (v[j] >= thr) ? v[j] : 0.0f);
    }
#pragma unroll
    for (int off = 16; off > 0; off >>= 1) mx = fmaxf(mx, __shfl_xor(mx, off, 32));
    float sum = 0.0f;
    for (int i = lane * 4; i < NKEY; i += 128) {
      v4f v = *(const v4f*)(Sr + i);
#pragma unroll
      for (int j = 0; j < 4; ++j) {
        const float mval = (v[j] >= thr) ? v[j] : 0.0f;
        const float e = __expf(mval - mx);
        v[j] = e;
        sum += e;
      }
      *(v4f*)(Sr + i) = v;
    }
#pragma unroll
    for (int off = 16; off > 0; off >>= 1) sum += __shfl_xor(sum, off, 32);
    if (lane == 0) invs[row] = 1.0f / sum;
  }
  __syncthreads();

  // ---- Phase 4: O = P @ V ; each wave owns 512 keys, partials in LDS ------
  v8f acc[4] = {};                                 // 4 column tiles of 16
  const int kstart = wave * 512;
  const float psc = invs[ln];
  const float* Pr = S + ln * NKEY;
  const __bf16* vb0 = gvt + (size_t)bat * DK * NKEY;  // Vt[64][4096]
  for (int kk = kstart; kk < kstart + 512; kk += 32) {
    const int ka = kk + half * 8;
    const int kc = kk + 16 + half * 8;
    v16bf A;
    {
      const v4f p0 = *(const v4f*)(Pr + ka);
      const v4f p1 = *(const v4f*)(Pr + ka + 4);
      const v4f p2 = *(const v4f*)(Pr + kc);
      const v4f p3 = *(const v4f*)(Pr + kc + 4);
#pragma unroll
      for (int j = 0; j < 4; ++j) {
        A[j]      = tobf(p0[j] * psc);
        A[4 + j]  = tobf(p1[j] * psc);
        A[8 + j]  = tobf(p2[j] * psc);
        A[12 + j] = tobf(p3[j] * psc);
      }
    }
#pragma unroll
    for (int nt = 0; nt < 4; ++nt) {
      const __bf16* vr = vb0 + (size_t)(nt * 16 + ln) * NKEY;  // column d
      const v16bf Bv = cat8(*(const v8bf*)(vr + ka), *(const v8bf*)(vr + kc));
      acc[nt] = wmma_bf16(A, Bv, acc[nt]);
    }
  }
  float* pp = partial + wave * (16 * 64);
#pragma unroll
  for (int r = 0; r < 8; ++r) {
    const int mm = r + 8 * half;
#pragma unroll
    for (int nt = 0; nt < 4; ++nt)
      pp[mm * 64 + nt * 16 + ln] = acc[nt][r];
  }
  __syncthreads();
  for (int idx = tid; idx < 16 * 64; idx += 256) {
    float s = 0.f;
#pragma unroll
    for (int w = 0; w < 8; ++w) s += partial[w * (16 * 64) + idx];
    const int mm = idx >> 6, nn = idx & 63;
    gao[(qrow0 + mm) * DK + nn] = tobf(s);
  }
}

// ---------------------------------------------------------------------------
// K3: out projection. One wave per 16x16 tile; K=64 in 2 WMMA steps.
// ---------------------------------------------------------------------------
__global__ __launch_bounds__(32)
void out_proj_kernel(const __bf16* __restrict__ gao, const float* __restrict__ Wout,
                     const float* __restrict__ bout, float* __restrict__ out)
{
  const int lane = threadIdx.x;
  const int row0 = blockIdx.x * 16;     // 16384/16
  const int col0 = blockIdx.y * 16;     // 256/16
  const int ln   = lane & 15;
  const int half = lane >> 4;

  v8f acc = {};
  const __bf16* ar = gao + (size_t)(row0 + ln) * DK;
#pragma unroll
  for (int s = 0; s < 2; ++s) {
    const int ka = s * 32 + half * 8;
    const int kc = s * 32 + 16 + half * 8;
    const v16bf A = cat8(*(const v8bf*)(ar + ka), *(const v8bf*)(ar + kc));
    v16bf Bv;
#pragma unroll
    for (int j = 0; j < 8; ++j) {
      Bv[j]     = tobf(Wout[(size_t)(ka + j) * DM + col0 + ln]);
      Bv[8 + j] = tobf(Wout[(size_t)(kc + j) * DM + col0 + ln]);
    }
    acc = wmma_bf16(A, Bv, acc);
  }
#pragma unroll
  for (int r = 0; r < 8; ++r) {
    const int mm = r + 8 * half;
    out[(size_t)(row0 + mm) * DM + col0 + ln] = acc[r] + bout[col0 + ln];
  }
}

// ---------------------------------------------------------------------------
extern "C" void kernel_launch(void* const* d_in, const int* in_sizes, int n_in,
                              void* d_out, int out_size, void* d_ws, size_t ws_size,
                              hipStream_t stream)
{
  const float* x    = (const float*)d_in[0];
  const float* Wqkv = (const float*)d_in[1];
  const float* bqkv = (const float*)d_in[2];
  const float* Wout = (const float*)d_in[3];
  const float* bout = (const float*)d_in[4];
  float* out = (float*)d_out;

  // ws layout: q | k | Vt | attn_out, all bf16, 2 MB each (8 MB total)
  __bf16* gq  = (__bf16*)d_ws;
  __bf16* gk  = gq  + (size_t)ROWS * DK;
  __bf16* gvt = gk  + (size_t)ROWS * DK;   // [4][64][4096]
  __bf16* gao = gvt + (size_t)ROWS * DK;

  qkv_proj_kernel<<<dim3(ROWS / 16, DQKV / 16), dim3(32), 0, stream>>>(
      x, Wqkv, bqkv, gq, gk, gvt);

  // Allow >64KB dynamic LDS (303 KB of the WGP's 320 KB).
  (void)hipFuncSetAttribute((const void*)attention_kernel,
                            hipFuncAttributeMaxDynamicSharedMemorySize,
                            (int)LDS_TOTAL);
  attention_kernel<<<dim3(ROWS / 16), dim3(256), LDS_TOTAL, stream>>>(
      gq, gk, gvt, gao);

  out_proj_kernel<<<dim3(ROWS / 16, DM / 16), dim3(32), 0, stream>>>(
      gao, Wout, bout, out);
}
